// LDBlock_14714557956528
// MI455X (gfx1250) — matrix operand order repaired and checked by reading, not compiled, not run on hardware
//
#include <hip/hip_runtime.h>
#include <cmath>

typedef __bf16 bf16;
typedef __attribute__((ext_vector_type(16))) __bf16 v16bf;
typedef __attribute__((ext_vector_type(8)))  __bf16 v8bf;
typedef __attribute__((ext_vector_type(8)))  float  v8f;
typedef __attribute__((ext_vector_type(4)))  unsigned int v4u;
typedef __attribute__((ext_vector_type(8)))  int v8i;
typedef __attribute__((ext_vector_type(4)))  int v4i;

#define B_    32
#define C_    256
#define H_    56
#define W_    56
#define NH_   8
#define LC_   7
#define DIL_  8
#define N_TOK 49
#define DH_   32
#define NWIN  2048            // B * DIL * DIL
#define MTOK  100352          // NWIN * N_TOK (divisible by 128)
#define SCALE_ 0.17677669529663687f   // DH^-0.5

// ---------------------------------------------------------------- helpers
static __device__ __forceinline__ v8f v8f_zero() {
  v8f z;
#pragma unroll
  for (int i = 0; i < 8; ++i) z[i] = 0.0f;
  return z;
}
static __device__ __forceinline__ v8bf v8bf_zero() {
  v8bf z;
#pragma unroll
  for (int i = 0; i < 8; ++i) z[i] = (bf16)0.0f;
  return z;
}
static __device__ __forceinline__ v16bf mk16(v8bf lo, v8bf hi) {
  return __builtin_shufflevector(lo, hi, 0,1,2,3,4,5,6,7,8,9,10,11,12,13,14,15);
}
static __device__ __forceinline__ v8f wmma_bf16(v16bf a, v16bf b, v8f c) {
  // D = A(16x32 bf16) x B(32x16 bf16) + C(16x16 f32)
  return __builtin_amdgcn_wmma_f32_16x16x32_bf16(false, a, false, b, (short)0, c, false, false);
}

// TDM: async 2-D tile load Global -> LDS (Tensor Data Mover, ISA ch.7/8).
// data_size = 2 bytes (bf16). tile0 = elems/row, tile1 = rows,
// stride/td0/td1 in elements. Issued once per wave (TDM ignores EXEC).
// NOTE: this toolchain exposes the 6-arg builtin (trailing int32x8 zero-filled).
static __device__ __forceinline__ void tdm_load_2d(unsigned lds_off, const void* g,
                                                   int tile0, int tile1, int stride,
                                                   int td0, int td1) {
  unsigned long long ga = (unsigned long long)(size_t)g;
  v4u g0;
  g0[0] = 1u;                                          // count=1, no gather, user D#
  g0[1] = lds_off;                                     // LDS byte address
  g0[2] = (unsigned)ga;                                // global_addr[31:0]
  g0[3] = (unsigned)((ga >> 32) & 0x1FFFFFFull) | (2u << 30);  // addr[56:32] | type=2
  v8i g1;
  g1[0] = (int)(1u << 16);                             // data_size = 1 -> 2 bytes
  g1[1] = (int)(((unsigned)td0 & 0xFFFFu) << 16);      // tensor_dim0[15:0]
  g1[2] = (int)((((unsigned)td0 >> 16) & 0xFFFFu) | (((unsigned)td1 & 0xFFFFu) << 16));
  g1[3] = (int)(((((unsigned)td1) >> 16) & 0xFFFFu) | (((unsigned)tile0 & 0xFFFFu) << 16));
  g1[4] = (int)((unsigned)tile1 & 0xFFFFu);            // tile_dim1, tile_dim2=0
  g1[5] = stride;                                      // tensor_dim0_stride[31:0]
  g1[6] = 0;
  g1[7] = 0;
  v4i gz = {0, 0, 0, 0};
  v8i gz8 = {0, 0, 0, 0, 0, 0, 0, 0};
  __builtin_amdgcn_tensor_load_to_lds(g0, g1, gz, gz, gz8, 0);
}

// ---------------------------------------------------------------- weight convert+transpose
// w: [K][N] f32 row-major  ->  wt: [N][K] bf16 row-major (B-operand layout)
__global__ void cvt_transpose(const float* __restrict__ w, bf16* __restrict__ wt,
                              int K, int N) {
  int idx = blockIdx.x * blockDim.x + threadIdx.x;
  int total = K * N;
  if (idx < total) {
    int n = idx / K;
    int kk = idx - n * K;
    wt[idx] = (bf16)w[kk * N + n];
  }
}

// ---------------------------------------------------------------- CPE + LN1 + window partition
__global__ void pre_kernel(const float* __restrict__ x, const float* __restrict__ cw,
                           const float* __restrict__ cb, const float* __restrict__ gam,
                           const float* __restrict__ bet, float* __restrict__ sc,
                           bf16* __restrict__ xln) {
  int p = blockIdx.x;
  int c = threadIdx.x;
  int b = p / (H_ * W_);
  int rem = p - b * H_ * W_;
  int h = rem / W_;
  int w = rem - h * W_;
  const size_t plane = (size_t)H_ * W_;
  const float* xc = x + ((size_t)b * C_ + c) * plane;
  float val = xc[h * W_ + w];
  float acc = 0.f;
#pragma unroll
  for (int dh = -1; dh <= 1; ++dh) {
#pragma unroll
    for (int dw = -1; dw <= 1; ++dw) {
      int hh = h + dh, ww = w + dw;
      if (hh >= 0 && hh < H_ && ww >= 0 && ww < W_)
        acc += xc[hh * W_ + ww] * cw[c * 9 + (dh + 1) * 3 + (dw + 1)];
    }
  }
  val = val + acc + cb[c];
  float s = val, ss = val * val;
#pragma unroll
  for (int m = 16; m >= 1; m >>= 1) { s += __shfl_xor(s, m, 32); ss += __shfl_xor(ss, m, 32); }
  __shared__ float sA[8], sB[8];
  int wv = threadIdx.x >> 5, lane = threadIdx.x & 31;
  if (lane == 0) { sA[wv] = s; sB[wv] = ss; }
  __syncthreads();
  float mean = 0.f, m2 = 0.f;
#pragma unroll
  for (int i = 0; i < 8; ++i) { mean += sA[i]; m2 += sB[i]; }
  mean *= (1.0f / C_); m2 *= (1.0f / C_);
  float rstd = rsqrtf(m2 - mean * mean + 1e-5f);
  float nrm = (val - mean) * rstd * gam[c] + bet[c];
  int l1 = h >> 3, d1 = h & 7, l2 = w >> 3, d2 = w & 7;
  int win = (b * 8 + d1) * 8 + d2;
  size_t t = (size_t)win * N_TOK + l1 * LC_ + l2;
  sc[t * C_ + c] = val;
  xln[t * C_ + c] = (bf16)nrm;
}

// ---------------------------------------------------------------- LN2 (token order f32 -> bf16)
__global__ void ln_kernel(const float* __restrict__ xin, const float* __restrict__ gam,
                          const float* __restrict__ bet, bf16* __restrict__ xo) {
  size_t t = blockIdx.x;
  int c = threadIdx.x;
  float val = xin[t * C_ + c];
  float s = val, ss = val * val;
#pragma unroll
  for (int m = 16; m >= 1; m >>= 1) { s += __shfl_xor(s, m, 32); ss += __shfl_xor(ss, m, 32); }
  __shared__ float sA[8], sB[8];
  int wv = threadIdx.x >> 5, lane = threadIdx.x & 31;
  if (lane == 0) { sA[wv] = s; sB[wv] = ss; }
  __syncthreads();
  float mean = 0.f, m2 = 0.f;
#pragma unroll
  for (int i = 0; i < 8; ++i) { mean += sA[i]; m2 += sB[i]; }
  mean *= (1.0f / C_); m2 *= (1.0f / C_);
  float rstd = rsqrtf(m2 - mean * mean + 1e-5f);
  xo[t * C_ + c] = (bf16)((val - mean) * rstd * gam[c] + bet[c]);
}

// ---------------------------------------------------------------- tiled WMMA GEMM (TDM-fed)
// out[M, N] = A[M, K](bf16) @ Bt[N, K]^T(bf16) + bias, per-mode epilogue.
// Block: 256 thr = 8 waves; tile 128(M) x 64(N); K step 32; LDS double-buffered,
// tiles streamed by the Tensor Data Mover (wave 0 issues, TENSORcnt paces).
template <int MODE>
__global__ void gemm_kernel(const bf16* __restrict__ A, const bf16* __restrict__ Bt,
                            const float* __restrict__ bias, int K,
                            float* __restrict__ resid,
                            bf16* __restrict__ oq, bf16* __restrict__ okk, bf16* __restrict__ ovv,
                            bf16* __restrict__ obf, float* __restrict__ ofin) {
  __shared__ __attribute__((aligned(64))) bf16 As[2][128][32];
  __shared__ __attribute__((aligned(64))) bf16 Bs[2][64][32];
  const int tid = threadIdx.x;
  const int wv = tid >> 5, lane = tid & 31, g2 = lane >> 4, r = lane & 15;
  const size_t M0 = (size_t)blockIdx.x * 128;
  const int n0 = blockIdx.y * 64;
  v8f acc[4];
#pragma unroll
  for (int t = 0; t < 4; ++t) acc[t] = v8f_zero();

  const unsigned ldsA = (unsigned)(size_t)&As[0][0][0];
  const unsigned ldsB = (unsigned)(size_t)&Bs[0][0][0];
  const int nk = K >> 5;

  if (tid < 32) {   // wave 0 drives the TDM
    tdm_load_2d(ldsA, &A[M0 * (size_t)K], 32, 128, K, K, 128);
    tdm_load_2d(ldsB, &Bt[(size_t)n0 * K], 32, 64, K, K, 64);
  }
  for (int s = 0; s < nk; ++s) {
    const int buf = s & 1;
    if (tid < 32) {
      if (s + 1 < nk) {
        int k1 = (s + 1) << 5;
        tdm_load_2d(ldsA + (buf ^ 1) * 128 * 32 * 2, &A[M0 * (size_t)K + k1], 32, 128, K, K, 128);
        tdm_load_2d(ldsB + (buf ^ 1) * 64 * 32 * 2, &Bt[(size_t)n0 * K + k1], 32, 64, K, K, 64);
        __builtin_amdgcn_s_wait_tensorcnt(2);  // current stage done, next in flight
      } else {
        __builtin_amdgcn_s_wait_tensorcnt(0);
      }
    }
    __syncthreads();
    int ar = wv * 16 + r;                       // A frag: contiguous K half by lane group
    v16bf av = mk16(*(const v8bf*)&As[buf][ar][16 * g2], *(const v8bf*)&As[buf][ar][16 * g2 + 8]);
#pragma unroll
    for (int t = 0; t < 4; ++t) {               // B frag: lane -> col, K half by lane group
      v16bf bv = *(const v16bf*)&Bs[buf][t * 16 + r][16 * g2];
      acc[t] = wmma_bf16(av, bv, acc[t]);
    }
    __syncthreads();                            // compute done before buf is re-filled
  }
  // epilogue: C/D layout -> row = rr + 8*(lane>>4), col = lane&15
#pragma unroll
  for (int rr = 0; rr < 8; ++rr) {
    size_t row = M0 + wv * 16 + rr + 8 * g2;
    size_t obase = 0;
    if (MODE == 3) {                            // token -> BCHW mapping, once per row
      int win = (int)(row / N_TOK);
      int n = (int)(row - (size_t)win * N_TOK);
      int d2 = win & 7, d1 = (win >> 3) & 7, bb = win >> 6;
      int l1 = n / LC_, l2 = n - l1 * LC_;
      int h = l1 * DIL_ + d1, w = l2 * DIL_ + d2;
      obase = ((size_t)bb * C_ * H_ + h) * W_ + w;
    }
#pragma unroll
    for (int t = 0; t < 4; ++t) {
      int col = n0 + t * 16 + r;
      float val = acc[t][rr] + bias[col];
      if (MODE == 0) {
        if (col < 256)      oq[row * 256 + col]          = (bf16)(val * SCALE_);
        else if (col < 512) okk[row * 256 + (col - 256)] = (bf16)val;
        else                ovv[row * 256 + (col - 512)] = (bf16)val;
      } else if (MODE == 1) {
        val += resid[row * 256 + col];
        resid[row * 256 + col] = val;
      } else if (MODE == 2) {
        val = 0.5f * val * (1.0f + erff(val * 0.70710678118654752f));
        obf[row * 1024 + col] = (bf16)val;
      } else {
        val += resid[row * 256 + col];
        ofin[obase + (size_t)col * (H_ * W_)] = val;
      }
    }
  }
}

// ---------------------------------------------------------------- windowed attention
// Block: 256 thr = 8 waves; one block per (window, head-pair). wave = (head_local, query-tile).
__global__ void attn_kernel(const bf16* __restrict__ q, const bf16* __restrict__ k,
                            const bf16* __restrict__ v, const float* __restrict__ rpb,
                            bf16* __restrict__ o) {
  __shared__ __attribute__((aligned(64))) bf16 Qs[2][64][32];
  __shared__ __attribute__((aligned(64))) bf16 Ks[2][64][32];
  __shared__ __attribute__((aligned(64))) bf16 Vts[2][32][64];
  __shared__ __attribute__((aligned(64))) bf16 Ps[8][16][64];
  __shared__ float rpbs[169 * NH_];
  const int tid = threadIdx.x;
  const int win = blockIdx.x >> 2, pair = blockIdx.x & 3;
  const size_t tw = (size_t)win * N_TOK;
  // ---- stage rel-pos bias table in LDS (branchless softmax later)
  for (int i = tid; i < 169 * NH_; i += 256) rpbs[i] = rpb[i];
  // ---- load Q/K rows (pad 49->64 with zeros)
  for (int u = tid; u < 512; u += 256) {
    int hl = u >> 8, rem = u & 255, n = rem >> 2, seg = rem & 3;
    int head = pair * 2 + hl;
    v8bf vq = v8bf_zero(), vk = v8bf_zero();
    if (n < N_TOK) {
      vq = *(const v8bf*)&q[(tw + n) * C_ + head * DH_ + seg * 8];
      vk = *(const v8bf*)&k[(tw + n) * C_ + head * DH_ + seg * 8];
    }
    *(v8bf*)&Qs[hl][n][seg * 8] = vq;
    *(v8bf*)&Ks[hl][n][seg * 8] = vk;
  }
  // ---- load V transposed: Vts[d][token]
  for (int e = tid; e < 4096; e += 256) {
    int hl = e >> 11, rem = e & 2047, n = rem >> 5, d = rem & 31;
    int head = pair * 2 + hl;
    bf16 val = (bf16)0.0f;
    if (n < N_TOK) val = v[(tw + n) * C_ + head * DH_ + d];
    Vts[hl][d][n] = val;
  }
  __syncthreads();
  const int wv = tid >> 5, lane = tid & 31, g2 = lane >> 4, r = lane & 15;
  const int hl = wv >> 2, mt = wv & 3;
  const int head = pair * 2 + hl;
  // ---- S = Q K^T (Q pre-scaled in QKV epilogue)
  int ar = mt * 16 + r;
  v16bf av = mk16(*(const v8bf*)&Qs[hl][ar][16 * g2], *(const v8bf*)&Qs[hl][ar][16 * g2 + 8]);
  v8f sacc[4];
#pragma unroll
  for (int t = 0; t < 4; ++t) sacc[t] = v8f_zero();
#pragma unroll
  for (int t = 0; t < 4; ++t) {
    v16bf bv = *(const v16bf*)&Ks[hl][t * 16 + r][16 * g2];
    sacc[t] = wmma_bf16(av, bv, sacc[t]);
  }
  // ---- bias + mask + softmax (row lives in 16 lanes of one half)
#pragma unroll
  for (int rr = 0; rr < 8; ++rr) {
    int n = mt * 16 + rr + 8 * g2;
    bool rowok = n < N_TOK;
    int nn = rowok ? n : 0;
    int l1 = nn / LC_, l2 = nn - l1 * LC_;
    float fv[4];
#pragma unroll
    for (int t = 0; t < 4; ++t) {
      int m = t * 16 + r;
      int q1 = m / LC_, q2 = m - q1 * LC_;
      bool ok = rowok && (m < N_TOK);
      int idx = ok ? ((l1 - q1 + 6) * 13 + (l2 - q2 + 6)) : 0;
      float sv = sacc[t][rr] + rpbs[idx * NH_ + head];
      fv[t] = ok ? sv : -1e30f;
    }
    float mx = fmaxf(fmaxf(fv[0], fv[1]), fmaxf(fv[2], fv[3]));
#pragma unroll
    for (int m = 8; m >= 1; m >>= 1) mx = fmaxf(mx, __shfl_xor(mx, m, 32));
    float pv[4]; float sum = 0.f;
#pragma unroll
    for (int t = 0; t < 4; ++t) { pv[t] = expf(fv[t] - mx); sum += pv[t]; }
#pragma unroll
    for (int m = 8; m >= 1; m >>= 1) sum += __shfl_xor(sum, m, 32);
    float inv = 1.0f / sum;
#pragma unroll
    for (int t = 0; t < 4; ++t) Ps[wv][rr + 8 * g2][t * 16 + r] = (bf16)(pv[t] * inv);
  }
  __syncthreads();
  // ---- O = P V  (K = 64 tokens in two 32-chunks, N = DH in two 16-tiles)
  v8f oacc[2];
#pragma unroll
  for (int d = 0; d < 2; ++d) oacc[d] = v8f_zero();
#pragma unroll
  for (int kc = 0; kc < 2; ++kc) {
    int kb = kc * 32;
    v16bf ap = mk16(*(const v8bf*)&Ps[wv][r][kb + 16 * g2],
                    *(const v8bf*)&Ps[wv][r][kb + 16 * g2 + 8]);
#pragma unroll
    for (int dt = 0; dt < 2; ++dt) {
      v16bf bv = *(const v16bf*)&Vts[hl][dt * 16 + r][kb + 16 * g2];
      oacc[dt] = wmma_bf16(ap, bv, oacc[dt]);
    }
  }
#pragma unroll
  for (int dt = 0; dt < 2; ++dt) {
#pragma unroll
    for (int rr = 0; rr < 8; ++rr) {
      int n = mt * 16 + rr + 8 * g2;
      if (n < N_TOK)
        o[(tw + n) * C_ + head * DH_ + dt * 16 + r] = (bf16)(oacc[dt][rr]);
    }
  }
}

// ---------------------------------------------------------------- launcher
extern "C" void kernel_launch(void* const* d_in, const int* in_sizes, int n_in,
                              void* d_out, int out_size, void* d_ws, size_t ws_size,
                              hipStream_t stream) {
  (void)in_sizes; (void)n_in; (void)out_size; (void)ws_size;
  const float* x      = (const float*)d_in[0];
  const float* cpe_w  = (const float*)d_in[1];
  const float* cpe_b  = (const float*)d_in[2];
  const float* ln1_w  = (const float*)d_in[3];
  const float* ln1_b  = (const float*)d_in[4];
  const float* rpb    = (const float*)d_in[5];
  const float* qkv_w  = (const float*)d_in[6];
  const float* qkv_b  = (const float*)d_in[7];
  const float* proj_w = (const float*)d_in[8];
  const float* proj_b = (const float*)d_in[9];
  const float* ln2_w  = (const float*)d_in[10];
  const float* ln2_b  = (const float*)d_in[11];
  const float* fc1_w  = (const float*)d_in[12];
  const float* fc1_b  = (const float*)d_in[13];
  const float* fc2_w  = (const float*)d_in[14];
  const float* fc2_b  = (const float*)d_in[15];

  char* ws = (char*)d_ws;
  size_t off = 0;
  float* sc1 = (float*)(ws + off); off += (size_t)MTOK * C_ * 4;   // shortcut / x2 (in-place)
  bf16* qb   = (bf16*)(ws + off);  off += (size_t)MTOK * C_ * 2;   // q (later: xln2)
  bf16* kb   = (bf16*)(ws + off);  off += (size_t)MTOK * C_ * 2;
  bf16* vb   = (bf16*)(ws + off);  off += (size_t)MTOK * C_ * 2;
  bf16* xb   = (bf16*)(ws + off);  off += (size_t)MTOK * C_ * 2;   // xln1 (later: attn out)
  bf16* h1   = (bf16*)(ws + off);  off += (size_t)MTOK * 1024 * 2;
  bf16* qkv_wt  = (bf16*)(ws + off); off += (size_t)768 * 256 * 2;
  bf16* proj_wt = (bf16*)(ws + off); off += (size_t)256 * 256 * 2;
  bf16* fc1_wt  = (bf16*)(ws + off); off += (size_t)1024 * 256 * 2;
  bf16* fc2_wt  = (bf16*)(ws + off); off += (size_t)256 * 1024 * 2;

  cvt_transpose<<<768, 256, 0, stream>>>(qkv_w, qkv_wt, 256, 768);
  cvt_transpose<<<256, 256, 0, stream>>>(proj_w, proj_wt, 256, 256);
  cvt_transpose<<<1024, 256, 0, stream>>>(fc1_w, fc1_wt, 256, 1024);
  cvt_transpose<<<1024, 256, 0, stream>>>(fc2_w, fc2_wt, 1024, 256);

  pre_kernel<<<MTOK, 256, 0, stream>>>(x, cpe_w, cpe_b, ln1_w, ln1_b, sc1, xb);
  gemm_kernel<0><<<dim3(784, 12), 256, 0, stream>>>(xb, qkv_wt, qkv_b, 256,
      nullptr, qb, kb, vb, nullptr, nullptr);
  attn_kernel<<<NWIN * 4, 256, 0, stream>>>(qb, kb, vb, rpb, xb);
  gemm_kernel<1><<<dim3(784, 4), 256, 0, stream>>>(xb, proj_wt, proj_b, 256,
      sc1, nullptr, nullptr, nullptr, nullptr, nullptr);
  ln_kernel<<<MTOK, 256, 0, stream>>>(sc1, ln2_w, ln2_b, qb);
  gemm_kernel<2><<<dim3(784, 16), 256, 0, stream>>>(qb, fc1_wt, fc1_b, 256,
      nullptr, nullptr, nullptr, nullptr, h1, nullptr);
  gemm_kernel<3><<<dim3(784, 4), 256, 0, stream>>>(h1, fc2_wt, fc2_b, 1024,
      sc1, nullptr, nullptr, nullptr, nullptr, (float*)d_out);
}
